// GraphLSTM_3599182594883
// MI455X (gfx1250) — matrix operand order repaired and checked by reading, not compiled, not run on hardware
//
#include <hip/hip_runtime.h>
#include <math.h>

// Problem constants (from the reference)
#define NN   4096       // nodes
#define EE   65536      // edges
#define DIN  64
#define HH   128
#define TT   8
#define BB   8
#define GG   512        // 4*H gate width
#define MROWS (BB*NN)   // 32768 GEMM rows

typedef __bf16 bf16;
typedef __attribute__((ext_vector_type(16))) __bf16 v16bf;
typedef __attribute__((ext_vector_type(8)))  __bf16 v8bf;
typedef __attribute__((ext_vector_type(8)))  float  v8f;

// ---------------------------------------------------------------- utilities
__global__ void k_zero_f32(float* __restrict__ p, long n) {
    long i = (long)blockIdx.x * blockDim.x + threadIdx.x;
    long stride = (long)gridDim.x * blockDim.x;
    for (; i < n; i += stride) p[i] = 0.0f;
}

__global__ void k_deg(const int* __restrict__ ei, float* __restrict__ deg) {
    int e = blockIdx.x * blockDim.x + threadIdx.x;
    if (e < EE) atomicAdd(&deg[ei[EE + e]], 1.0f);
}

__global__ void k_dinv(const float* __restrict__ deg, float* __restrict__ dinv,
                       float* __restrict__ selfnorm) {
    int n = blockIdx.x * blockDim.x + threadIdx.x;
    if (n < NN) {
        float d = deg[n] + 2.0f;       // improved=True: A + 2I
        float r = rsqrtf(d);
        dinv[n] = r;
        selfnorm[n] = 2.0f * r * r;    // self-loop weight 2 * dinv^2
    }
}

__global__ void k_norm_e(const int* __restrict__ ei, const float* __restrict__ dinv,
                         float* __restrict__ norm_e) {
    int e = blockIdx.x * blockDim.x + threadIdx.x;
    if (e < EE) norm_e[e] = dinv[ei[e]] * dinv[ei[EE + e]];
}

// W (K, 512) fp32 row-major  ->  Wt (512, K) bf16 row-major (contiguous in K)
__global__ void k_cast_wt(const float* __restrict__ W, bf16* __restrict__ Wt, int K) {
    int idx = blockIdx.x * blockDim.x + threadIdx.x;
    int total = K * GG;
    if (idx < total) {
        int g = idx % GG;
        int k = idx / GG;
        Wt[(long)g * K + k] = (bf16)W[idx];
    }
}

// comb row = [ x(D) | h(H) ] cast to bf16.  row = b*NN + n.
__global__ __launch_bounds__(256)
void k_pack(bf16* __restrict__ comb, const float* __restrict__ xsrc,
            long xBatchStride, int D, int K, const float* __restrict__ h) {
    int row = blockIdx.x;
    int b = row >> 12;            // NN = 4096
    int n = row & (NN - 1);
    int tid = threadIdx.x;
    if (tid < K) {
        float v;
        if (tid < D) v = xsrc[(long)b * xBatchStride + (long)n * D + tid];
        else         v = h[(long)row * HH + (tid - D)];
        comb[(long)row * K + tid] = (bf16)v;
    }
}

// ---------------------------------------------------------------- WMMA GEMM
// xw(M, 512) = comb(M, K) @ W(K, 512), bf16 inputs, f32 accumulate.
// Wave: 16-row M tile x 128-col strip (8 sub-tiles of 16), K-loop step 32.
// A fragment layout (ISA 7.12.2, 16-bit A 16x32): lane<16 -> K {0..7,16..23},
// lane>=16 -> K {8..15,24..31}.  B fragment: lane = column, same K order as A
// (lanes 0-15 K {0..7,16..23}, lanes 16-31 K {8..15,24..31}; Wt is (G,K)).
__global__ __launch_bounds__(256)
void k_gemm_wmma(const bf16* __restrict__ comb, const bf16* __restrict__ Wt,
                 float* __restrict__ xw, int K) {
    const int lane  = threadIdx.x & 31;
    const int wave  = threadIdx.x >> 5;
    const int m0    = (blockIdx.x * 8 + wave) * 16;
    const int g0    = blockIdx.y * 128;
    const int khalf = lane >> 4;
    const int l15   = lane & 15;

    v8f acc[8];
    #pragma unroll
    for (int i = 0; i < 8; ++i) acc[i] = (v8f){0.f,0.f,0.f,0.f,0.f,0.f,0.f,0.f};

    const bf16* arow = comb + (long)(m0 + l15) * K + khalf * 8;
    const int kTiles = K >> 5;

    for (int kt = 0; kt < kTiles; ++kt) {
        const bf16* pa = arow + kt * 32;
        v8bf alo = *(const v8bf*)(pa);
        v8bf ahi = *(const v8bf*)(pa + 16);
        v16bf a;
        #pragma unroll
        for (int i = 0; i < 8; ++i) { a[i] = alo[i]; a[8 + i] = ahi[i]; }

        #pragma unroll
        for (int nt = 0; nt < 8; ++nt) {
            const int col = g0 + nt * 16 + l15;
            const bf16* pb = Wt + (long)col * K + kt * 32 + khalf * 8;
            v8bf blo = *(const v8bf*)(pb);
            v8bf bhi = *(const v8bf*)(pb + 16);
            v16bf bm;
            #pragma unroll
            for (int i = 0; i < 8; ++i) { bm[i] = blo[i]; bm[8 + i] = bhi[i]; }
            acc[nt] = __builtin_amdgcn_wmma_f32_16x16x32_bf16(
                false, a, false, bm, (short)0, acc[nt], false, false);
        }
    }

    // C/D layout: VGPR r -> M = m0 + r + 8*khalf ; N = col
    #pragma unroll
    for (int nt = 0; nt < 8; ++nt) {
        const int col = g0 + nt * 16 + l15;
        #pragma unroll
        for (int r = 0; r < 8; ++r) {
            int row = m0 + r + 8 * khalf;
            xw[(long)row * GG + col] = acc[nt][r];
        }
    }
}

// agg = selfnorm[n] * xw + bias   (fused self-loop + bias)
__global__ __launch_bounds__(256)
void k_init_agg(float* __restrict__ agg, const float* __restrict__ xw,
                const float* __restrict__ selfnorm, const float* __restrict__ bias) {
    long total = (long)MROWS * GG;
    long i = (long)blockIdx.x * blockDim.x + threadIdx.x;
    long stride = (long)gridDim.x * blockDim.x;
    for (; i < total; i += stride) {
        int g    = (int)(i & (GG - 1));
        int node = (int)((i >> 9) & (NN - 1));
        agg[i] = selfnorm[node] * xw[i] + bias[g];
    }
}

// agg[b, dst, :] += norm_e * xw[b, src, :]   one block per (edge, batch)
__global__ __launch_bounds__(128)
void k_edge_scatter(const int* __restrict__ ei, const float* __restrict__ norm_e,
                    const float* __restrict__ xw, float* __restrict__ agg) {
    int e = blockIdx.x;
    int b = blockIdx.y;
    int src = ei[e];
    int dst = ei[EE + e];
    float w = norm_e[e];
    const float4* sp = (const float4*)(xw + ((long)b * NN + src) * GG);
    float*        dp = agg + ((long)b * NN + dst) * GG;
    float4 v = sp[threadIdx.x];
    int g = threadIdx.x * 4;
    atomicAdd(dp + g + 0, w * v.x);
    atomicAdd(dp + g + 1, w * v.y);
    atomicAdd(dp + g + 2, w * v.z);
    atomicAdd(dp + g + 3, w * v.w);
}

// LSTM cell update; writes h into state and into the layer output buffer.
__global__ __launch_bounds__(128)
void k_lstm(const float* __restrict__ agg, float* __restrict__ h, float* __restrict__ c,
            float* __restrict__ out, long outBatchStride) {
    int row = blockIdx.x;                 // b*NN + n
    int j = threadIdx.x;                  // 0..127
    const float* g = agg + (long)row * GG;
    float ci = g[j], cf = g[j + HH], co = g[j + 2 * HH], cg = g[j + 3 * HH];
    float i = 1.0f / (1.0f + __expf(-ci));
    float f = 1.0f / (1.0f + __expf(-cf));
    float o = 1.0f / (1.0f + __expf(-co));
    float gt = tanhf(cg);
    long idx = (long)row * HH + j;
    float cn = f * c[idx] + i * gt;
    float hn = o * tanhf(cn);
    c[idx] = cn;
    h[idx] = hn;
    int b = row >> 12;
    int n = row & (NN - 1);
    out[(long)b * outBatchStride + (long)n * HH + j] = hn;
}

// ---------------------------------------------------------------- launcher
extern "C" void kernel_launch(void* const* d_in, const int* in_sizes, int n_in,
                              void* d_out, int out_size, void* d_ws, size_t ws_size,
                              hipStream_t stream) {
    const float* x  = (const float*)d_in[0];   // (B,T,N,64)
    const float* W0 = (const float*)d_in[1];   // (192,512)
    const float* b0 = (const float*)d_in[2];   // (512)
    const float* W1 = (const float*)d_in[3];   // (256,512)
    const float* b1 = (const float*)d_in[4];   // (512)
    const int*   ei = (const int*)d_in[5];     // (2,E)
    float* out = (float*)d_out;                // (B,T,N,128)

    // Workspace carve-up
    char* ws = (char*)d_ws;
    size_t off = 0;
    auto alloc = [&](size_t bytes) {
        void* p = ws + off;
        off += (bytes + 255) & ~(size_t)255;
        return p;
    };
    float* deg      = (float*)alloc((size_t)NN * 4);
    float* dinv     = (float*)alloc((size_t)NN * 4);
    float* selfnorm = (float*)alloc((size_t)NN * 4);
    float* norm_e   = (float*)alloc((size_t)EE * 4);
    bf16*  W0t      = (bf16*)alloc((size_t)GG * 192 * 2);
    bf16*  W1t      = (bf16*)alloc((size_t)GG * 256 * 2);
    bf16*  comb     = (bf16*)alloc((size_t)MROWS * 256 * 2);
    float* xw       = (float*)alloc((size_t)MROWS * GG * 4);
    float* agg      = (float*)alloc((size_t)MROWS * GG * 4);
    float* h0       = (float*)alloc((size_t)MROWS * HH * 4);
    float* c0       = (float*)alloc((size_t)MROWS * HH * 4);
    float* h1       = (float*)alloc((size_t)MROWS * HH * 4);
    float* c1       = (float*)alloc((size_t)MROWS * HH * 4);
    float* hs0t     = (float*)alloc((size_t)MROWS * HH * 4);
    (void)ws_size; (void)in_sizes; (void)n_in; (void)out_size;

    // ---- setup: graph norm + weight cast + state init
    k_zero_f32<<<64, 256, 0, stream>>>(deg, NN);
    k_deg<<<EE / 256, 256, 0, stream>>>(ei, deg);
    k_dinv<<<NN / 256, 256, 0, stream>>>(deg, dinv, selfnorm);
    k_norm_e<<<EE / 256, 256, 0, stream>>>(ei, dinv, norm_e);
    k_cast_wt<<<(192 * GG + 255) / 256, 256, 0, stream>>>(W0, W0t, 192);
    k_cast_wt<<<(256 * GG + 255) / 256, 256, 0, stream>>>(W1, W1t, 256);
    k_zero_f32<<<2048, 256, 0, stream>>>(h0, (long)MROWS * HH);
    k_zero_f32<<<2048, 256, 0, stream>>>(c0, (long)MROWS * HH);
    k_zero_f32<<<2048, 256, 0, stream>>>(h1, (long)MROWS * HH);
    k_zero_f32<<<2048, 256, 0, stream>>>(c1, (long)MROWS * HH);

    dim3 gemmGrid(MROWS / (16 * 8), GG / 128);   // 256 x 4 blocks, 8 waves each
    dim3 edgeGrid(EE, BB);

    for (int t = 0; t < TT; ++t) {
        // ---------------- layer 0: input = x[:, t], K = 64+128 = 192
        k_pack<<<MROWS, 256, 0, stream>>>(comb, x + (long)t * NN * DIN,
                                          (long)TT * NN * DIN, DIN, 192, h0);
        k_gemm_wmma<<<gemmGrid, 256, 0, stream>>>(comb, W0t, xw, 192);
        k_init_agg<<<8192, 256, 0, stream>>>(agg, xw, selfnorm, b0);
        k_edge_scatter<<<edgeGrid, 128, 0, stream>>>(ei, norm_e, xw, agg);
        k_lstm<<<MROWS, 128, 0, stream>>>(agg, h0, c0, hs0t, (long)NN * HH);

        // ---------------- layer 1: input = hs0t, K = 128+128 = 256
        k_pack<<<MROWS, 256, 0, stream>>>(comb, hs0t, (long)NN * HH, HH, 256, h1);
        k_gemm_wmma<<<gemmGrid, 256, 0, stream>>>(comb, W1t, xw, 256);
        k_init_agg<<<8192, 256, 0, stream>>>(agg, xw, selfnorm, b1);
        k_edge_scatter<<<edgeGrid, 128, 0, stream>>>(ei, norm_e, xw, agg);
        // output slice for time t: out + t*N*H, batch stride T*N*H
        k_lstm<<<MROWS, 128, 0, stream>>>(agg, h1, c1, out + (long)t * NN * HH,
                                          (long)TT * NN * HH);
    }
}
